// MambaBlock_14723147891395
// MI455X (gfx1250) — matrix operand order repaired and checked by reading, not compiled, not run on hardware
//
#include <hip/hip_runtime.h>

// ---------------- problem constants (from reference) ----------------
#define Bsz     2
#define Lseq    2048
#define DMODEL  1024
#define DSTATE  16
#define DCONV   4
#define DINNER  2048
#define DTRANK  64
#define ROWS    (Bsz * Lseq)          // 4096
#define N1      (2 * DINNER)          // 4096
#define NXP     (DTRANK + 2 * DSTATE) // 96
#define NXPP    128                   // NXP padded to GEMM tile multiple
#define LN_EPS  1e-5f

// GEMM block tiling: 128 threads = 4 waves; block tile 128x64;
// wave tile 32x64 (2 M-subtiles x 4 N-subtiles -> 24 WMMA / K-step / wave)
#define BM 128
#define BN 64
#define BK 32

typedef __attribute__((ext_vector_type(16))) __bf16 v16bf;
typedef __attribute__((ext_vector_type(8)))  float  v8f;

// fp32 -> bf16 hi + bf16 residual (bf16x2 split precision)
__device__ __forceinline__ void split2(float v, __bf16& hi, __bf16& lo) {
  __bf16 h = (__bf16)v;
  hi = h;
  lo = (__bf16)(v - (float)h);
}

// ---- CDNA5 async copy: 16B global -> LDS, tracked by ASYNCcnt ----
__device__ __forceinline__ void async_ld16(const void* lds_ptr, const void* gptr) {
  unsigned lds = (unsigned)(uintptr_t)lds_ptr;            // low 32b = LDS offset
  unsigned long long ga = (unsigned long long)(uintptr_t)gptr;
  asm volatile("global_load_async_to_lds_b128 %0, %1, off"
               :: "v"(lds), "v"(ga) : "memory");
}
// previous stage drained, next stage's 12 copies still in flight
__device__ __forceinline__ void wait_async_prev() {
  asm volatile("s_wait_asynccnt 0xc" ::: "memory");
}
__device__ __forceinline__ void wait_async_0() {
  asm volatile("s_wait_asynccnt 0x0" ::: "memory");
}

// ---- LDS fragment loads (layouts per CDNA5 ISA 7.12.2) ----
// A 16x32 bf16: lane m(<16)=row m; k = half*8+{0..7} and 16+half*8+{0..7}
__device__ __forceinline__ v16bf lds_a_frag(const __bf16* As, int row, int half) {
  union { uint4 u[2]; v16bf v; } cv;
  cv.u[0] = *(const uint4*)(As + row * BK + half * 8);
  cv.u[1] = *(const uint4*)(As + row * BK + 16 + half * 8);
  return cv.v;
}
// B 32x16 bf16: lane n(<16)=column n; k = half*8+{0..7} and 16+half*8+{0..7}
__device__ __forceinline__ v16bf lds_b_frag(const __bf16* Bs, int krow, int col) {
  v16bf v;
  int h = krow >> 4, n = col + (krow & 15);
#pragma unroll
  for (int s = 0; s < 16; ++s) {
    int k = (s < 8) ? (h * 8 + s) : (16 + h * 8 + (s - 8));
    v[s] = Bs[k * BN + n];
  }
  return v;
}

// stage one BMxBK A (hi/lo) + BKxBN B (hi/lo) block into LDS via async engine:
// 12 async ops per thread (=> ASYNCcnt +12 per wave per stage)
__device__ __forceinline__ void stage_tile(
    const __bf16* Ah, const __bf16* Al, const __bf16* Bh, const __bf16* Bl,
    int N, int K, int rowBase, int colBase, int k0,
    __bf16* AsH, __bf16* AsL, __bf16* BsH, __bf16* BsL, int tid) {
  int ar = tid >> 2, as = (tid & 3) * 8;     // A: 128 rows x 4 x 16B segs
#pragma unroll
  for (int rr = 0; rr < 4; ++rr) {
    int row = ar + rr * 32;
    async_ld16(&AsH[row * BK + as], Ah + (size_t)(rowBase + row) * K + k0 + as);
    async_ld16(&AsL[row * BK + as], Al + (size_t)(rowBase + row) * K + k0 + as);
  }
  int br = tid >> 3, bs = (tid & 7) * 8;     // B: 32 rows x 8 x 16B segs
#pragma unroll
  for (int rr = 0; rr < 2; ++rr) {
    int row = br + rr * 16;
    async_ld16(&BsH[row * BN + bs], Bh + (size_t)(k0 + row) * N + colBase + bs);
    async_ld16(&BsL[row * BN + bs], Bl + (size_t)(k0 + row) * N + colBase + bs);
  }
}

// one K-step of compute: 4 A-frag + 8 B-frag LDS reads, 24 WMMAs
__device__ __forceinline__ void compute_step(
    const __bf16* AsH, const __bf16* AsL, const __bf16* BsH, const __bf16* BsL,
    int wrow, int lane, int half, v8f (&acc)[2][4]) {
  v16bf a_hi[2], a_lo[2];
#pragma unroll
  for (int mi = 0; mi < 2; ++mi) {
    a_hi[mi] = lds_a_frag(AsH, wrow + mi * 16 + (lane & 15), half);
    a_lo[mi] = lds_a_frag(AsL, wrow + mi * 16 + (lane & 15), half);
  }
#pragma unroll
  for (int nt = 0; nt < 4; ++nt) {
    v16bf b_hi = lds_b_frag(BsH, lane, nt * 16);
    v16bf b_lo = lds_b_frag(BsL, lane, nt * 16);
#pragma unroll
    for (int mi = 0; mi < 2; ++mi) {
      // bf16x2: hi*hi + hi*lo + lo*hi  (fp32 accumulate)
      acc[mi][nt] = __builtin_amdgcn_wmma_f32_16x16x32_bf16(false, a_hi[mi], false, b_hi,
                                                            (short)0, acc[mi][nt], false, false);
      acc[mi][nt] = __builtin_amdgcn_wmma_f32_16x16x32_bf16(false, a_hi[mi], false, b_lo,
                                                            (short)0, acc[mi][nt], false, false);
      acc[mi][nt] = __builtin_amdgcn_wmma_f32_16x16x32_bf16(false, a_lo[mi], false, b_hi,
                                                            (short)0, acc[mi][nt], false, false);
    }
  }
}

// ---------------- LDS-tiled WMMA GEMM, bf16x2, fp32 acc --------------------
__global__ __launch_bounds__(128) void k_gemm_lds(
    const __bf16* __restrict__ Ah, const __bf16* __restrict__ Al,
    const __bf16* __restrict__ Bh, const __bf16* __restrict__ Bl,
    float* __restrict__ C, int M, int N, int K) {
  __shared__ __align__(16) __bf16 AsH[2][BM * BK], AsL[2][BM * BK];
  __shared__ __align__(16) __bf16 BsH[2][BK * BN], BsL[2][BK * BN];

  int tid  = threadIdx.x;
  int wave = tid >> 5, lane = tid & 31;
  int half = lane >> 4;
  int rowBase = blockIdx.y * BM;
  int colBase = blockIdx.x * BN;
  int wrow = wave * 32;                     // wave covers rows [wrow, wrow+32)

  v8f acc[2][4] = {{{}, {}, {}, {}}, {{}, {}, {}, {}}};

  int nsteps = K / BK;                      // >= 2 for all calls
  stage_tile(Ah, Al, Bh, Bl, N, K, rowBase, colBase, 0,
             AsH[0], AsL[0], BsH[0], BsL[0], tid);

  for (int ks = 0; ks < nsteps - 1; ++ks) {
    int buf = ks & 1;
    stage_tile(Ah, Al, Bh, Bl, N, K, rowBase, colBase, (ks + 1) * BK,
               AsH[buf ^ 1], AsL[buf ^ 1], BsH[buf ^ 1], BsL[buf ^ 1], tid);
    wait_async_prev();
    __syncthreads();
    compute_step(AsH[buf], AsL[buf], BsH[buf], BsL[buf], wrow, lane, half, acc);
    __syncthreads();
  }
  // epilogue: last staged buffer
  wait_async_0();
  __syncthreads();
  {
    int buf = (nsteps - 1) & 1;
    compute_step(AsH[buf], AsL[buf], BsH[buf], BsL[buf], wrow, lane, half, acc);
  }

  // C/D layout: VGPR i, lanes 0-15 -> M=i, lanes 16-31 -> M=8+i; N = lane%16
  int n  = lane & 15;
  int mb = half * 8;
#pragma unroll
  for (int mi = 0; mi < 2; ++mi)
#pragma unroll
    for (int nt = 0; nt < 4; ++nt)
#pragma unroll
      for (int i = 0; i < 8; ++i)
        C[(size_t)(rowBase + wrow + mi * 16 + mb + i) * N + colBase + nt * 16 + n] =
            acc[mi][nt][i];
}

// ---------------- LayerNorm + bf16x2 split of xn ----------------
__global__ void k_ln_split(const float* __restrict__ x,
                           const float* __restrict__ g,
                           const float* __restrict__ beta,
                           __bf16* __restrict__ xhi, __bf16* __restrict__ xlo) {
  __shared__ float sm[256], sv[256];
  int row = blockIdx.x;
  const float* xr = x + (size_t)row * DMODEL;
  float s = 0.f, s2 = 0.f;
  for (int i = threadIdx.x; i < DMODEL; i += 256) {
    float v = xr[i]; s += v; s2 += v * v;
  }
  sm[threadIdx.x] = s; sv[threadIdx.x] = s2;
  __syncthreads();
  for (int o = 128; o > 0; o >>= 1) {
    if ((int)threadIdx.x < o) {
      sm[threadIdx.x] += sm[threadIdx.x + o];
      sv[threadIdx.x] += sv[threadIdx.x + o];
    }
    __syncthreads();
  }
  float mu  = sm[0] * (1.0f / DMODEL);
  float var = sv[0] * (1.0f / DMODEL) - mu * mu;
  float rs  = rsqrtf(var + LN_EPS);
  for (int i = threadIdx.x; i < DMODEL; i += 256) {
    float v = (xr[i] - mu) * rs * g[i] + beta[i];
    __bf16 h, l; split2(v, h, l);
    xhi[(size_t)row * DMODEL + i] = h;
    xlo[(size_t)row * DMODEL + i] = l;
  }
}

// ---------------- generic fp32 -> bf16 hi/lo ----------------
__global__ void k_split(const float* __restrict__ a, __bf16* __restrict__ hi,
                        __bf16* __restrict__ lo, int n) {
  int i = blockIdx.x * 256 + threadIdx.x;
  if (i < n) { __bf16 h, l; split2(a[i], h, l); hi[i] = h; lo[i] = l; }
}

// fp32 [rows,cols] -> bf16 hi/lo [rows,colsPad], zero-padded columns
__global__ void k_split_pad(const float* __restrict__ a, __bf16* __restrict__ hi,
                            __bf16* __restrict__ lo, int rows, int cols, int colsPad) {
  int i = blockIdx.x * 256 + threadIdx.x;
  if (i < rows * colsPad) {
    int r = i / colsPad, c = i % colsPad;
    float v = (c < cols) ? a[(size_t)r * cols + c] : 0.f;
    __bf16 h, l; split2(v, h, l);
    hi[i] = h; lo[i] = l;
  }
}

// split dt columns (0..63) of dbc[ROWS, NXPP] into compact [ROWS, 64]
__global__ void k_split_dt(const float* __restrict__ dbc,
                           __bf16* __restrict__ hi, __bf16* __restrict__ lo) {
  int i = blockIdx.x * 256 + threadIdx.x;
  if (i < ROWS * DTRANK) {
    int r = i / DTRANK, c = i % DTRANK;
    __bf16 h, l; split2(dbc[(size_t)r * NXPP + c], h, l);
    hi[i] = h; lo[i] = l;
  }
}

// ---------------- causal depthwise conv(k=4) + bias + SiLU ----------------
__global__ void k_conv_silu(const float* __restrict__ C1,
                            const float* __restrict__ cw, const float* __restrict__ cb,
                            float* __restrict__ u,
                            __bf16* __restrict__ uhi, __bf16* __restrict__ ulo) {
  int i = blockIdx.x * 256 + threadIdx.x;
  if (i >= ROWS * DINNER) return;
  int d   = i % DINNER;
  int row = i / DINNER;
  int l   = row % Lseq;
  int b   = row / Lseq;
  float acc = cb[d];
#pragma unroll
  for (int k = 0; k < DCONV; ++k) {
    int t = l + k - (DCONV - 1);
    if (t >= 0)
      acc += C1[(size_t)(b * Lseq + t) * N1 + d] * cw[d * DCONV + k];
  }
  float su = acc / (1.0f + __expf(-acc));
  u[i] = su;
  __bf16 h, lo; split2(su, h, lo);
  uhi[i] = h; ulo[i] = lo;
}

// ---------------- delta = softplus(dt@W_dt + b_dt), in place ----------------
__global__ void k_delta(float* __restrict__ Cdt, const float* __restrict__ bdt) {
  int i = blockIdx.x * 256 + threadIdx.x;
  if (i < ROWS * DINNER) {
    int d = i % DINNER;
    float v = Cdt[i] + bdt[d];
    Cdt[i] = (v > 20.f) ? v : log1pf(__expf(v));
  }
}

// ---------------- selective scan, fused with D-skip and silu(z) gate --------
__global__ void k_scan(const float* __restrict__ u, const float* __restrict__ delta,
                       const float* __restrict__ dbc,   // [ROWS, NXPP]
                       const float* __restrict__ A_log, const float* __restrict__ Dsk,
                       const float* __restrict__ C1,    // z at col DINNER+d
                       __bf16* __restrict__ yhi, __bf16* __restrict__ ylo) {
  int idx = blockIdx.x * 256 + threadIdx.x;
  if (idx >= Bsz * DINNER) return;
  int b = idx >> 11;
  int d = idx & (DINNER - 1);
  float A[DSTATE];
#pragma unroll
  for (int s = 0; s < DSTATE; ++s) A[s] = -__expf(A_log[d * DSTATE + s]);
  float dskip = Dsk[d];
  float h[DSTATE];
#pragma unroll
  for (int s = 0; s < DSTATE; ++s) h[s] = 0.f;

  for (int t = 0; t < Lseq; ++t) {
    size_t r   = (size_t)b * Lseq + t;
    float  ut  = u[r * DINNER + d];
    float  dl  = delta[r * DINNER + d];
    const float* bc = dbc + r * NXPP;
    float du = dl * ut;
    float y  = 0.f;
#pragma unroll
    for (int s = 0; s < DSTATE; ++s) {
      float dA = __expf(dl * A[s]);
      h[s] = dA * h[s] + du * bc[DTRANK + s];
      y   += h[s] * bc[DTRANK + DSTATE + s];
    }
    y += ut * dskip;
    float z  = C1[r * N1 + DINNER + d];
    float yz = y * (z / (1.0f + __expf(-z)));
    __bf16 hh, ll; split2(yz, hh, ll);
    yhi[r * DINNER + d] = hh;
    ylo[r * DINNER + d] = ll;
  }
}

// ---------------- residual add ----------------
__global__ void k_add(const float* __restrict__ x, const float* __restrict__ C3,
                      float* __restrict__ out) {
  int i = blockIdx.x * 256 + threadIdx.x;
  if (i < ROWS * DMODEL) out[i] = x[i] + C3[i];
}

// =========================== host launcher ===========================
extern "C" void kernel_launch(void* const* d_in, const int* in_sizes, int n_in,
                              void* d_out, int out_size, void* d_ws, size_t ws_size,
                              hipStream_t stream) {
  const float* x      = (const float*)d_in[0];
  const float* ln_g   = (const float*)d_in[1];
  const float* ln_b   = (const float*)d_in[2];
  const float* W_in   = (const float*)d_in[3];
  const float* conv_w = (const float*)d_in[4];
  const float* conv_b = (const float*)d_in[5];
  const float* W_xp   = (const float*)d_in[6];
  const float* W_dt   = (const float*)d_in[7];
  const float* b_dt   = (const float*)d_in[8];
  const float* A_log  = (const float*)d_in[9];
  const float* D_skip = (const float*)d_in[10];
  const float* W_out  = (const float*)d_in[11];
  float* out = (float*)d_out;

  size_t off = 0;
  auto carve = [&](size_t bytes) -> void* {
    void* p = (char*)d_ws + off;
    off += (bytes + 255) & ~(size_t)255;
    return p;
  };
  // weights, bf16x2 (W_xp zero-padded to NXPP cols)
  __bf16* Win_hi  = (__bf16*)carve((size_t)DMODEL * N1 * 2);
  __bf16* Win_lo  = (__bf16*)carve((size_t)DMODEL * N1 * 2);
  __bf16* Wxp_hi  = (__bf16*)carve((size_t)DINNER * NXPP * 2);
  __bf16* Wxp_lo  = (__bf16*)carve((size_t)DINNER * NXPP * 2);
  __bf16* Wdt_hi  = (__bf16*)carve((size_t)DTRANK * DINNER * 2);
  __bf16* Wdt_lo  = (__bf16*)carve((size_t)DTRANK * DINNER * 2);
  __bf16* Wout_hi = (__bf16*)carve((size_t)DINNER * DMODEL * 2);
  __bf16* Wout_lo = (__bf16*)carve((size_t)DINNER * DMODEL * 2);
  // activations
  __bf16* xn_hi = (__bf16*)carve((size_t)ROWS * DMODEL * 2);
  __bf16* xn_lo = (__bf16*)carve((size_t)ROWS * DMODEL * 2);
  float*  C1    = (float*) carve((size_t)ROWS * N1 * 4);      // [u_pre | z]
  float*  u     = (float*) carve((size_t)ROWS * DINNER * 4);
  __bf16* u_hi  = (__bf16*)carve((size_t)ROWS * DINNER * 2);
  __bf16* u_lo  = (__bf16*)carve((size_t)ROWS * DINNER * 2);
  float*  dbc   = (float*) carve((size_t)ROWS * NXPP * 4);
  __bf16* dt_hi = (__bf16*)carve((size_t)ROWS * DTRANK * 2);
  __bf16* dt_lo = (__bf16*)carve((size_t)ROWS * DTRANK * 2);
  float*  delta = (float*) carve((size_t)ROWS * DINNER * 4);
  __bf16* y_hi  = (__bf16*)carve((size_t)ROWS * DINNER * 2);
  __bf16* y_lo  = (__bf16*)carve((size_t)ROWS * DINNER * 2);
  float*  C3    = (float*) carve((size_t)ROWS * DMODEL * 4);
  (void)ws_size; (void)in_sizes; (void)n_in; (void)out_size;

  auto blocks = [](size_t n) { return (unsigned)((n + 255) / 256); };
  auto gemm = [&](const __bf16* Ah, const __bf16* Al, const __bf16* Bh,
                  const __bf16* Bl, float* C, int M, int N, int K) {
    dim3 grid(N / BN, M / BM);
    k_gemm_lds<<<grid, 128, 0, stream>>>(Ah, Al, Bh, Bl, C, M, N, K);
  };

  // 1) LayerNorm + split
  k_ln_split<<<ROWS, 256, 0, stream>>>(x, ln_g, ln_b, xn_hi, xn_lo);
  // 2) weight splits
  k_split<<<blocks((size_t)DMODEL * N1), 256, 0, stream>>>(W_in, Win_hi, Win_lo, DMODEL * N1);
  k_split_pad<<<blocks((size_t)DINNER * NXPP), 256, 0, stream>>>(W_xp, Wxp_hi, Wxp_lo,
                                                                 DINNER, NXP, NXPP);
  k_split<<<blocks((size_t)DTRANK * DINNER), 256, 0, stream>>>(W_dt, Wdt_hi, Wdt_lo, DTRANK * DINNER);
  k_split<<<blocks((size_t)DINNER * DMODEL), 256, 0, stream>>>(W_out, Wout_hi, Wout_lo, DINNER * DMODEL);
  // 3) xn @ W_in -> [u_pre | z]
  gemm(xn_hi, xn_lo, Win_hi, Win_lo, C1, ROWS, N1, DMODEL);
  // 4) depthwise conv + SiLU
  k_conv_silu<<<blocks((size_t)ROWS * DINNER), 256, 0, stream>>>(C1, conv_w, conv_b, u, u_hi, u_lo);
  // 5) u @ W_xp -> dbc (padded)
  gemm(u_hi, u_lo, Wxp_hi, Wxp_lo, dbc, ROWS, NXPP, DINNER);
  // 6) dt split; dt @ W_dt; softplus
  k_split_dt<<<blocks((size_t)ROWS * DTRANK), 256, 0, stream>>>(dbc, dt_hi, dt_lo);
  gemm(dt_hi, dt_lo, Wdt_hi, Wdt_lo, delta, ROWS, DINNER, DTRANK);
  k_delta<<<blocks((size_t)ROWS * DINNER), 256, 0, stream>>>(delta, b_dt);
  // 7) selective scan fused with D-skip + silu(z) gate
  k_scan<<<(Bsz * DINNER) / 256, 256, 0, stream>>>(u, delta, dbc, A_log, D_skip, C1, y_hi, y_lo);
  // 8) y @ W_out, residual add
  gemm(y_hi, y_lo, Wout_hi, Wout_lo, C3, ROWS, DMODEL, DINNER);
  k_add<<<blocks((size_t)ROWS * DMODEL), 256, 0, stream>>>(x, C3, out);
}
